// SAGE_30210799960810
// MI455X (gfx1250) — matrix operand order repaired and checked by reading, not compiled, not run on hardware
//
#include <hip/hip_runtime.h>
#include <hip/hip_bf16.h>

typedef __attribute__((ext_vector_type(16))) _Float16 v16h;
typedef __attribute__((ext_vector_type(8)))  float    v8f;

#define D_IN  50
#define NH1   36
#define N_CLS 40
#define KPAD  64      // padded K for both layers (>=50, >=36), mult of 32
#define CPAD  48      // padded out-cols for both layers (>=36, >=40), 3 tiles

// ---------------------------------------------------------------- utilities
__global__ void sage_zero_f32(float* __restrict__ p, long long n) {
    long long i = (long long)blockIdx.x * blockDim.x + threadIdx.x;
    if (i < n) p[i] = 0.0f;
}

// copy [n x D] row-major into zero-padded [n x KPAD]
__global__ void sage_pad_copy(const float* __restrict__ s, float* __restrict__ d,
                              long long n, int D) {
    long long t = (long long)blockIdx.x * blockDim.x + threadIdx.x;
    if (t >= n * KPAD) return;
    long long r = t >> 6;
    int f = (int)(t & (KPAD - 1));
    d[t] = (f < D) ? s[r * D + f] : 0.0f;
}

// W [K x C] row-major f32 -> Wt [CPAD x KPAD] f16 (transposed, zero padded)
__global__ void sage_prep_w(const float* __restrict__ W, _Float16* __restrict__ Wt,
                            int K, int C) {
    int t = blockIdx.x * blockDim.x + threadIdx.x;
    if (t >= CPAD * KPAD) return;
    int col = t >> 6, k = t & (KPAD - 1);
    Wt[t] = (_Float16)((col < C && k < K) ? W[k * C + col] : 0.0f);
}

// one thread per edge: in-degree counts
__global__ void sage_count(const int* __restrict__ dst, float* __restrict__ cnt, int E) {
    int e = blockIdx.x * blockDim.x + threadIdx.x;
    if (e < E) atomicAdd(&cnt[dst[e]], 1.0f);
}

// one thread per (edge, feature): acc[dst] += feat[src]; stride-64 rows, L2-resident
__global__ void sage_scatter(const float* __restrict__ feat,
                             const int* __restrict__ src,
                             const int* __restrict__ dst,
                             float* __restrict__ acc, int E, int D) {
    long long t = (long long)blockIdx.x * blockDim.x + threadIdx.x;
    int e = (int)(t / D);
    int f = (int)(t - (long long)e * D);
    if (e < E)
        atomicAdd(&acc[((long long)dst[e] << 6) + f],
                  feat[((long long)src[e] << 6) + f]);
}

// A fragment (16x32 f16): lane half selects K+16; element i ->
// K = kb*32 + i + half*16.  Row is stride-64, zero padded,
// so the fragment is 4 aligned float4 loads.
__device__ __forceinline__ v16h load_a_frag(const float* __restrict__ rowp,
                                            int kb, int half, float scale) {
    const float4* p = (const float4*)rowp;
    const int b = kb * 8 + half * 4;
    float4 a0 = p[b], a1 = p[b + 1], a2 = p[b + 2], a3 = p[b + 3];
    v16h r;
    r[0] = (_Float16)(a0.x * scale);  r[1] = (_Float16)(a0.y * scale);
    r[2] = (_Float16)(a0.z * scale);  r[3] = (_Float16)(a0.w * scale);
    r[4] = (_Float16)(a1.x * scale);  r[5] = (_Float16)(a1.y * scale);
    r[6] = (_Float16)(a1.z * scale);  r[7] = (_Float16)(a1.w * scale);
    r[8] = (_Float16)(a2.x * scale);  r[9] = (_Float16)(a2.y * scale);
    r[10] = (_Float16)(a2.z * scale); r[11] = (_Float16)(a2.w * scale);
    r[12] = (_Float16)(a3.x * scale); r[13] = (_Float16)(a3.y * scale);
    r[14] = (_Float16)(a3.z * scale); r[15] = (_Float16)(a3.w * scale);
    return r;
}

// ------------------------------------------------- fused SAGE layer via WMMA
// out[n x outStride] cols [0,C): relu?((agg/max(cnt,1))@Wl + bias + xr@Wr)
// agg/xr are stride-KPAD zero-padded; Wlt/Wrt are [CPAD x KPAD] f16 transposed.
// One wave per 16x16 tile; tile predicates wave-uniform -> EXEC all-1 at WMMA.
__global__ void sage_wmma_gemm(const float* __restrict__ agg,
                               const float* __restrict__ cnt,
                               const float* __restrict__ xr,
                               const _Float16* __restrict__ Wlt,
                               const _Float16* __restrict__ Wrt,
                               const float* __restrict__ bias,
                               float* __restrict__ out, int outStride,
                               int n, int C, int relu) {
    const int lane = threadIdx.x & 31;
    const int wave = threadIdx.x >> 5;
    const int nColTiles = CPAD / 16;                          // 3
    const int nRowTiles = (n + 15) >> 4;
    const int tile = blockIdx.x * (blockDim.x >> 5) + wave;   // wave-uniform
    if (tile >= nColTiles * nRowTiles) return;                // uniform branch
    const int tm = tile / nColTiles;
    const int tn = tile - tm * nColTiles;

    const int half = lane >> 4;
    const int l16  = lane & 15;
    const int row  = tm * 16 + l16;
    const int rowc = row < n ? row : n - 1;
    const int col  = tn * 16 + l16;                           // < CPAD always
    const int colc = col < C ? col : C - 1;

    const float inv = 1.0f / fmaxf(cnt[rowc], 1.0f);          // mean factor
    const float* aggRow = agg + ((long long)rowc << 6);
    const float* xrRow  = xr  + ((long long)rowc << 6);

    v8f c = {};
#pragma unroll
    for (int kb = 0; kb < KPAD / 32; ++kb) {                  // 2 iterations
        v16h aA = load_a_frag(aggRow, kb, half, inv);
        v16h aX = load_a_frag(xrRow,  kb, half, 1.0f);
        // B fragment: K = kb*32 + i + half*16 -> 16 consecutive f16, 32B aligned
        v16h bL = *(const v16h*)(Wlt + ((long long)col << 6) + kb * 32 + half * 16);
        v16h bR = *(const v16h*)(Wrt + ((long long)col << 6) + kb * 32 + half * 16);
        c = __builtin_amdgcn_wmma_f32_16x16x32_f16(false, aA, false, bL,
                                                   (short)0, c, false, false);
        c = __builtin_amdgcn_wmma_f32_16x16x32_f16(false, aX, false, bR,
                                                   (short)0, c, false, false);
    }

    const float bv = bias[colc];
    const bool colOK = (col < C);
#pragma unroll
    for (int i = 0; i < 8; ++i) {               // C/D layout: VGPR i -> M=i+half*8
        int r = tm * 16 + i + half * 8;
        float v = c[i] + bv;
        if (relu) v = fmaxf(v, 0.0f);
        if (colOK && r < n) out[(long long)r * outStride + col] = v;
    }
}

// --------------------------------------------------------------------- launch
extern "C" void kernel_launch(void* const* d_in, const int* in_sizes, int n_in,
                              void* d_out, int out_size, void* d_ws, size_t ws_size,
                              hipStream_t stream) {
    const float* x   = (const float*)d_in[0];
    const int*   ei  = (const int*)  d_in[1];
    const float* W1l = (const float*)d_in[2];
    const float* b1  = (const float*)d_in[3];
    const float* W1r = (const float*)d_in[4];
    const float* W2l = (const float*)d_in[5];
    const float* b2  = (const float*)d_in[6];
    const float* W2r = (const float*)d_in[7];

    const long long N = in_sizes[0] / D_IN;
    const int E = in_sizes[1] / 2;
    const int* src = ei;
    const int* dst = ei + E;

    // workspace layout (256B-aligned chunks)
    char* base = (char*)d_ws;
    size_t off = 0;
    auto take = [&](size_t bytes) { size_t o = off; off = (off + bytes + 255) & ~(size_t)255; return o; };
    float*    aggb = (float*)   (base + take((size_t)N * KPAD * 4)); // stride-64 accum
    float*    cnt  = (float*)   (base + take((size_t)N * 4));
    float*    hbuf = (float*)   (base + take((size_t)N * KPAD * 4)); // layer-1 output, padded
    size_t zeroBytesEnd = off;                                       // zero everything above
    float*    xpad = (float*)   (base + take((size_t)N * KPAD * 4)); // padded x
    _Float16* w1lt = (_Float16*)(base + take(CPAD * KPAD * 2));
    _Float16* w1rt = (_Float16*)(base + take(CPAD * KPAD * 2));
    _Float16* w2lt = (_Float16*)(base + take(CPAD * KPAD * 2));
    _Float16* w2rt = (_Float16*)(base + take(CPAD * KPAD * 2));

    const int B = 256;
    // 1) zero agg + cnt + hbuf (one sweep; covers padding columns)
    {
        long long nz = (long long)(zeroBytesEnd / 4);
        sage_zero_f32<<<(unsigned)((nz + B - 1) / B), B, 0, stream>>>((float*)base, nz);
    }
    // 2) pad x into stride-64 buffer
    {
        long long nt = N * KPAD;
        sage_pad_copy<<<(unsigned)((nt + B - 1) / B), B, 0, stream>>>(x, xpad, N, D_IN);
    }
    // 3) weight prep: f16, transposed, padded (tiny)
    {
        int nt = CPAD * KPAD, g = (nt + B - 1) / B;
        sage_prep_w<<<g, B, 0, stream>>>(W1l, w1lt, D_IN, NH1);
        sage_prep_w<<<g, B, 0, stream>>>(W1r, w1rt, D_IN, NH1);
        sage_prep_w<<<g, B, 0, stream>>>(W2l, w2lt, NH1, N_CLS);
        sage_prep_w<<<g, B, 0, stream>>>(W2r, w2rt, NH1, N_CLS);
    }
    // 4) in-degree counts
    sage_count<<<(E + B - 1) / B, B, 0, stream>>>(dst, cnt, E);
    // 5) scatter x into agg
    {
        long long nt = (long long)E * D_IN;
        sage_scatter<<<(unsigned)((nt + B - 1) / B), B, 0, stream>>>(xpad, src, dst, aggb, E, D_IN);
    }
    // 6) layer 1: hbuf = relu(mean(agg)@W1l + b1 + x@W1r)   [stride-64 out]
    {
        int tiles = (int)((N + 15) / 16) * (CPAD / 16);
        int wpb = B / 32;
        sage_wmma_gemm<<<(tiles + wpb - 1) / wpb, B, 0, stream>>>(
            aggb, cnt, xpad, w1lt, w1rt, b1, hbuf, KPAD, (int)N, NH1, 1);
    }
    // 7) re-zero agg for layer 2
    {
        long long nz = N * KPAD;
        sage_zero_f32<<<(unsigned)((nz + B - 1) / B), B, 0, stream>>>(aggb, nz);
    }
    // 8) scatter h into agg
    {
        long long nt = (long long)E * NH1;
        sage_scatter<<<(unsigned)((nt + B - 1) / B), B, 0, stream>>>(hbuf, src, dst, aggb, E, NH1);
    }
    // 9) layer 2: d_out = mean(agg)@W2l + b2 + h@W2r   [stride-40 out]
    {
        int tiles = (int)((N + 15) / 16) * (CPAD / 16);
        int wpb = B / 32;
        sage_wmma_gemm<<<(tiles + wpb - 1) / wpb, B, 0, stream>>>(
            aggb, cnt, hbuf, w2lt, w2rt, b2, (float*)d_out, N_CLS, (int)N, N_CLS, 0);
    }
}
